// CausalSelfAttention_46961172415255
// MI455X (gfx1250) — matrix operand order repaired and checked by reading, not compiled, not run on hardware
//
#include <hip/hip_runtime.h>
#include <hip/hip_bf16.h>
#include <stdint.h>

// ---- problem constants (match reference) ----
#define SEQ     2048
#define N_EMBD  2048
#define N_HEAD  16
#define HS      128          // head size
#define BATCH   2

#define QTILE   16           // q rows per wave
#define WAVES   8            // waves per block (256 threads, wave32)
#define BLOCK_QROWS (QTILE * WAVES)   // 128
#define KCHUNK  32           // k rows per chunk (one WMMA K=32 sweep of att)

// LDS strides padded for bank-conflict avoidance (must stay even for b32 loads)
#define KSTRIDE 34           // sK: [d][row], d-major (transposed K chunk)
#define VSTRIDE 132          // sV: [row][d], row-major
#define ASTRIDE 34           // sAtt per-wave scratch: [m][k]

typedef __attribute__((ext_vector_type(16))) __bf16 v16bf;
typedef __attribute__((ext_vector_type(8)))  float  v8f;

union BFVec { v16bf v; uint32_t u[8]; };
union FVec8 { v8f v; float f[8]; };

__device__ __forceinline__ uint16_t f2bf(float x) {
  uint32_t u = __float_as_uint(x);
  u += 0x7FFFu + ((u >> 16) & 1u);       // round-to-nearest-even
  return (uint16_t)(u >> 16);
}

// Pack two f32 -> packed bf16x2 (gfx1250 V_CVT_PK_BF16_F32 when available).
__device__ __forceinline__ uint32_t packbf(float lo, float hi) {
#if __has_builtin(__builtin_amdgcn_cvt_pk_bf16_f32)
  typedef __bf16 bf2 __attribute__((ext_vector_type(2)));
  union { bf2 v; uint32_t u; } r;
  r.v = __builtin_amdgcn_cvt_pk_bf16_f32(lo, hi);
  return r.u;
#else
  return (uint32_t)f2bf(lo) | ((uint32_t)f2bf(hi) << 16);
#endif
}

// gfx1250 hardware tanh (TRANS32 op).
__device__ __forceinline__ float fast_tanh(float x) {
#if __has_builtin(__builtin_amdgcn_tanhf)
  return __builtin_amdgcn_tanhf(x);
#else
  float y;
  asm volatile("v_tanh_f32 %0, %1\n\tv_nop" : "=v"(y) : "v"(x));
  return y;
#endif
}

// ---------------- pre-pass: per-(b,h,d) column sum of V ----------------
// out[q] = sum_{k<=q}(tanh(s)+1) v_k  -  sum_all v_k   (since tanh(-inf) = -1)
// This 16KB table lets the main loop stop at the causal diagonal (~2x work cut).
__global__ __launch_bounds__(128, 1)
void vsum_kernel(const float* __restrict__ V, float* __restrict__ Vsum) {
  const int d   = threadIdx.x;   // 0..127
  const int h   = blockIdx.x;    // 0..15
  const int bat = blockIdx.y;    // 0..1
  const float* vp = V + (size_t)bat * SEQ * N_EMBD + h * HS + d;
  float s0 = 0.f, s1 = 0.f, s2 = 0.f, s3 = 0.f;
  #pragma unroll 4
  for (int t = 0; t < SEQ; t += 4) {
    s0 += vp[(size_t)(t + 0) * N_EMBD];
    s1 += vp[(size_t)(t + 1) * N_EMBD];
    s2 += vp[(size_t)(t + 2) * N_EMBD];
    s3 += vp[(size_t)(t + 3) * N_EMBD];
  }
  Vsum[(bat * N_HEAD + h) * HS + d] = (s0 + s1) + (s2 + s3);
}

__global__ __launch_bounds__(256, 1)
void tanh_attn_rope_kernel(const float* __restrict__ Q,
                           const float* __restrict__ K,
                           const float* __restrict__ V,
                           const float* __restrict__ Vsum,
                           float* __restrict__ Out) {
  __shared__ uint16_t sK[HS * KSTRIDE];             // bf16, transposed [d][row]
  __shared__ uint16_t sV[KCHUNK * VSTRIDE];         // bf16, row-major [row][d]
  __shared__ uint16_t sAtt[WAVES][QTILE * ASTRIDE]; // per-wave C->A layout bounce

  const int lane = threadIdx.x & 31;
  const int wave = threadIdx.x >> 5;
  const int qblk = blockIdx.x;   // 0..15
  const int h    = blockIdx.y;   // 0..15
  const int bat  = blockIdx.z;   // 0..1

  const int qRow0 = qblk * BLOCK_QROWS + wave * QTILE;
  const int m     = lane & 15;   // WMMA M / N position within a 16-wide tile
  const int half  = lane >> 4;
  const int qRow  = qRow0 + m;

  const float scale = 0.08838834764831845f;   // 1/sqrt(128)
  const float lg10k = 9.210340371976184f;     // ln(10000)

  // ---------- Q tile -> WMMA A layout (RoPE fused into fp32->bf16) ----------
  BFVec qA[4];
  {
    const float* qp = Q + ((size_t)bat * SEQ + qRow) * N_EMBD + h * HS;
    #pragma unroll
    for (int c = 0; c < 4; ++c) {
      #pragma unroll
      for (int v = 0; v < 8; ++v) {
        const int kb = c * 32 + (v & 3) * 2 + half * 8 + (v >> 2) * 16;  // even
        float2 qe = *(const float2*)(qp + kb);
        const int i = kb >> 1;                 // RoPE pair index 0..63
        const float freq = __expf(-(float)i * (1.0f / 64.0f) * lg10k);
        float sa, ca;
        __sincosf((float)qRow * freq, &sa, &ca);
        qA[c].u[v] = packbf(qe.x * ca - qe.y * sa, qe.x * sa + qe.y * ca);
      }
    }
  }

  // ---------- RoPE recurrence state for the K-chunk cooperative loads ------
  float cA[8], sA[8], cS[8], sS[8];
  #pragma unroll
  for (int j = 0; j < 8; ++j) {
    const int p   = threadIdx.x + 256 * j;
    const int row = p >> 6;
    const int pr  = p & 63;
    const float freq = __expf(-(float)pr * (1.0f / 64.0f) * lg10k);
    __sincosf((float)row * freq, &sA[j], &cA[j]);          // angle at kRow0 = 0
    __sincosf((float)KCHUNK * freq, &sS[j], &cS[j]);       // per-chunk step
  }

  // ---------- loop-invariant LDS operand base pointers (per lane) ----------
  const uint16_t* sKb[4];
  #pragma unroll
  for (int c = 0; c < 4; ++c) sKb[c] = &sK[(c * 32) * KSTRIDE + m];
  const uint16_t* sVb[8];
  #pragma unroll
  for (int g = 0; g < 8; ++g) sVb[g] = &sV[g * 16 + m];
  const uint16_t* sAb = &sAtt[wave][m * ASTRIDE];          // A-layout reads
  uint16_t*       sAw = &sAtt[wave][(8 * half) * ASTRIDE]; // C-layout writes

  FVec8 O[8] = {};   // 16x128 fp32 accumulator tile (8 x 16x16 C tiles)

  // Causal cut: block only sweeps chunks up to its diagonal; each wave stops
  // at its own 16-row tile's diagonal chunk.
  const int nCh = qblk * 4 + 4;              // chunks 0 .. qblk*4+3
  const int dch = qblk * 4 + (wave >> 1);    // last chunk this wave computes

  for (int ch = 0; ch < nCh; ++ch) {
    const int kRow0 = ch * KCHUNK;

    // ---------- cooperative load of K (RoPE, transposed) and V chunks ----------
    #pragma unroll
    for (int j = 0; j < 8; ++j) {
      const int p   = threadIdx.x + 256 * j;  // pair index 0..2047
      const int row = p >> 6;                 // 0..31
      const int d   = (p & 63) * 2;
      const size_t base = ((size_t)bat * SEQ + (kRow0 + row)) * N_EMBD + h * HS + d;
      float2 ke = *(const float2*)(K + base);
      const float ca = cA[j], sa = sA[j];
      const uint32_t kp = packbf(ke.x * ca - ke.y * sa, ke.x * sa + ke.y * ca);
      sK[(d    ) * KSTRIDE + row] = (uint16_t)kp;
      sK[(d + 1) * KSTRIDE + row] = (uint16_t)(kp >> 16);
      // rotate RoPE phase forward by KCHUNK positions for the next chunk
      cA[j] = ca * cS[j] - sa * sS[j];
      sA[j] = sa * cS[j] + ca * sS[j];
      float2 ve = *(const float2*)(V + base);
      *(uint32_t*)&sV[row * VSTRIDE + d] = packbf(ve.x, ve.y);
      if (j == 0 && ch + 1 < nCh) {           // hint next chunk into cache
        __builtin_prefetch(K + base + (size_t)KCHUNK * N_EMBD, 0, 1);
        __builtin_prefetch(V + base + (size_t)KCHUNK * N_EMBD, 0, 1);
      }
    }
    __syncthreads();

    if (ch <= dch) {
      // ---------- scores: S(16x32) = Qtile(16x128) @ Kchunk(32x128)^T ----------
      v8f S0 = {}, S1 = {};
      #pragma unroll
      for (int c = 0; c < 4; ++c) {
        BFVec b0, b1;
        #pragma unroll
        for (int v = 0; v < 8; ++v) {
          const int kb = (v & 3) * 2 + half * 8 + (v >> 2) * 16;
          b0.u[v] = (uint32_t)sKb[c][kb * KSTRIDE] | ((uint32_t)sKb[c][(kb + 1) * KSTRIDE] << 16);
          b1.u[v] = (uint32_t)sKb[c][kb * KSTRIDE + 16] | ((uint32_t)sKb[c][(kb + 1) * KSTRIDE + 16] << 16);
        }
        S0 = __builtin_amdgcn_wmma_f32_16x16x32_bf16(false, qA[c].v, false, b0.v,
                                                     (short)0, S0, false, false);
        S1 = __builtin_amdgcn_wmma_f32_16x16x32_bf16(false, qA[c].v, false, b1.v,
                                                     (short)0, S1, false, false);
      }

      // ---------- att' = tanh(S*scale)+1 (diagonal chunk masks k>q to 0) -----
      {
        FVec8 s0, s1; s0.v = S0; s1.v = S1;
        const int n = lane & 15;
        if (ch == dch) {
          // only the diagonal chunk needs the causal predicate
          #pragma unroll
          for (int r = 0; r < 8; ++r) {
            const int qg = qRow0 + r + 8 * half;
            const float a0 = (kRow0 + n      > qg) ? 0.0f : fast_tanh(s0.f[r] * scale) + 1.0f;
            const float a1 = (kRow0 + 16 + n > qg) ? 0.0f : fast_tanh(s1.f[r] * scale) + 1.0f;
            const uint32_t ap = packbf(a0, a1);
            sAw[r * ASTRIDE + n     ] = (uint16_t)ap;
            sAw[r * ASTRIDE + 16 + n] = (uint16_t)(ap >> 16);
          }
        } else {
          // interior chunks: whole tile satisfies k <= q, no predicate
          #pragma unroll
          for (int r = 0; r < 8; ++r) {
            const float a0 = fast_tanh(s0.f[r] * scale) + 1.0f;
            const float a1 = fast_tanh(s1.f[r] * scale) + 1.0f;
            const uint32_t ap = packbf(a0, a1);
            sAw[r * ASTRIDE + n     ] = (uint16_t)ap;
            sAw[r * ASTRIDE + 16 + n] = (uint16_t)(ap >> 16);
          }
        }
      }
      BFVec attA;
      #pragma unroll
      for (int v = 0; v < 8; ++v) {
        const int kb = (v & 3) * 2 + half * 8 + (v >> 2) * 16;
        attA.u[v] = *(const uint32_t*)(sAb + kb);
      }

      // ---------- O(16x128) += att'(16x32) @ Vchunk(32x128) ----------
      // Ping-pong two operand buffers so two LDS fetches stay in flight and the
      // scheduler can use partial s_wait_dscnt instead of draining per WMMA.
      BFVec vbA, vbB;
      #pragma unroll
      for (int v = 0; v < 8; ++v) { const int kb = (v & 3) * 2 + half * 8 + (v >> 2) * 16; vbA.u[v] = (uint32_t)sVb[0][kb * VSTRIDE] | ((uint32_t)sVb[0][(kb + 1) * VSTRIDE] << 16); }
      #pragma unroll
      for (int g = 0; g < 8; ++g) {
        BFVec& cur = (g & 1) ? vbB : vbA;
        BFVec& nxt = (g & 1) ? vbA : vbB;
        if (g + 1 < 8) {
          #pragma unroll
          for (int v = 0; v < 8; ++v) {
            const int kb = (v & 3) * 2 + half * 8 + (v >> 2) * 16;
            nxt.u[v] = (uint32_t)sVb[g + 1][kb * VSTRIDE] | ((uint32_t)sVb[g + 1][(kb + 1) * VSTRIDE] << 16);
          }
        }
        O[g].v = __builtin_amdgcn_wmma_f32_16x16x32_bf16(false, attA.v, false, cur.v,
                                                         (short)0, O[g].v, false, false);
      }
    }
    __syncthreads();
  }

  // ---------- store fp32 output: Out = O - colsum(V) (mask correction) ------
  {
    const int n = lane & 15;
    const float* vs = Vsum + (bat * N_HEAD + h) * HS;
    #pragma unroll
    for (int g = 0; g < 8; ++g) {
      const float sub = vs[g * 16 + n];
      #pragma unroll
      for (int r = 0; r < 8; ++r) {
        const int mm = r + 8 * half;
        Out[((size_t)bat * SEQ + (qRow0 + mm)) * N_EMBD + h * HS + g * 16 + n] =
            O[g].f[r] - sub;
      }
    }
  }
}

extern "C" void kernel_launch(void* const* d_in, const int* in_sizes, int n_in,
                              void* d_out, int out_size, void* d_ws, size_t ws_size,
                              hipStream_t stream) {
  (void)in_sizes; (void)n_in; (void)out_size; (void)ws_size;
  const float* Q = (const float*)d_in[0];
  const float* K = (const float*)d_in[1];
  const float* V = (const float*)d_in[2];
  float* Out  = (float*)d_out;
  float* Vsum = (float*)d_ws;                    // B*H*HS floats = 16 KB

  dim3 vgrid(N_HEAD, BATCH);                     // (16, 2)
  vsum_kernel<<<vgrid, dim3(HS), 0, stream>>>(V, Vsum);

  dim3 grid(SEQ / BLOCK_QROWS, N_HEAD, BATCH);   // (16, 16, 2)
  dim3 block(256);                               // 8 waves of 32
  tanh_attn_rope_kernel<<<grid, block, 0, stream>>>(Q, K, V, Vsum, Out);
}
